// CrossAttention_50070728737272
// MI455X (gfx1250) — matrix operand order repaired and checked by reading, not compiled, not run on hardware
//
#include <hip/hip_runtime.h>
#include <hip/hip_bf16.h>
#include <stdint.h>

// ---------------------------------------------------------------------------
// CDNA5 (gfx1250) cross-attention: bf16 WMMA, f32 accumulate, TDM + async DMA.
//   D=1024, H=16, HD=64, B=8, T=1024, N=1536
// Pipeline:
//   0) convert query/key_value/W* f32 -> bf16 (bandwidth-trivial prep pass)
//   1-3) Q/K/V projections: bf16 GEMM, double-buffered TENSOR_LOAD_TO_LDS
//   4) flash attention per (b,h,128-row tile), async-LDS K staging -> AO
//   5) out = AO @ Wo^T + bo (f32 out)
// Workspace: 64M u16 elems = 128 MB.
// ---------------------------------------------------------------------------

typedef unsigned short u16;
typedef __attribute__((ext_vector_type(16))) __bf16 v16bf;
typedef __attribute__((ext_vector_type(8)))  float  v8f;
typedef __attribute__((ext_vector_type(8)))  unsigned short v8u;   // 16 B
typedef __attribute__((ext_vector_type(16))) unsigned short v16u;  // 32 B
typedef __attribute__((ext_vector_type(4)))  unsigned int v4u;     // SGPR quad
typedef __attribute__((ext_vector_type(8)))  unsigned int v8ui;    // SGPR octet

#define DEV __device__ __forceinline__

DEV u16 bf16_of(float f) {              // hardware f32 -> bf16 (RNE)
  union { __bf16 b; u16 s; } u;
  u.b = (__bf16)f;
  return u.s;
}
DEV void st_out(float* p, float v) { *p = v; }
DEV void st_out(u16* p,   float v) { *p = bf16_of(v); }

DEV v8f wmma_bf16(v16bf a, v16bf b, v8f c) {
  return __builtin_amdgcn_wmma_f32_16x16x32_bf16(false, a, false, b,
                                                 (short)0, c, false, false);
}

// --- per-lane async global->LDS DMA (ASYNCcnt), ISA 10.7/15.18 -------------
// Low 32 bits of a generic pointer to __shared__ are the DS byte address.
DEV void async_copy16(u16* lds_dst, const u16* gsrc) {
  const uint32_t ldsa = (uint32_t)(uintptr_t)lds_dst;
  asm volatile("global_load_async_to_lds_b128 %0, %1, off"
               :: "v"(ldsa), "v"(gsrc) : "memory");
}
DEV void async_wait0() {
  asm volatile("s_wait_asynccnt 0x0" ::: "memory");
}

// --- Tensor Data Mover: 2D tile load, descriptor per ISA 8.3/8.4 -----------
// Groups 0/1 only (VADDR2/3 omitted => tensors up to 2D).
// tile_d0 elems contiguous per row, tile_d1 rows, row stride in elems, 2B/elem.
// LDS layout: tile row-major contiguous at lds_byte_addr.
DEV void tdm_load_2d_bf16(const u16* gsrc, uint32_t lds_byte_addr,
                          uint32_t tile_d0, uint32_t tile_d1,
                          uint32_t stride_elems) {
  const uint64_t ga = (uint64_t)(uintptr_t)gsrc;
  v4u g0;
  g0[0] = 1u;                                   // count=1, user descriptor
  g0[1] = lds_byte_addr;                        // lds_addr [63:32]
  g0[2] = (uint32_t)ga;                         // global_addr [95:64]
  g0[3] = (uint32_t)((ga >> 32) & 0x01FFFFFFu)  // global_addr [120:96]
          | (2u << 30);                         // type=2 ("image") [127:126]
  v8ui g1;
  g1[0] = 1u << 16;                             // wg_mask=0, data_size=1 (2B)
  g1[1] = (tile_d0 & 0xFFFFu) << 16;            // tensor_dim0[15:0] @ [63:48]
  g1[2] = ((tile_d0 >> 16) & 0xFFFFu)           // tensor_dim0[31:16]
          | ((tile_d1 & 0xFFFFu) << 16);        // tensor_dim1[15:0] @ [111:80]
  g1[3] = ((tile_d1 >> 16) & 0xFFFFu)           // tensor_dim1[31:16]
          | ((tile_d0 & 0xFFFFu) << 16);        // tile_dim0 @ [127:112]
  g1[4] = tile_d1 & 0xFFFFu;                    // tile_dim1 @ [143:128]
  g1[5] = stride_elems;                         // tensor_dim0_stride[31:0]
  g1[6] = 0u;                                   // stride[47:32], dim1_stride lo
  g1[7] = 0u;                                   // dim1_stride hi (unused, 2D)
  asm volatile("tensor_load_to_lds %0, %1" :: "s"(g0), "s"(g1) : "memory");
}
DEV void tensor_wait0() {
#if defined(__has_builtin) && __has_builtin(__builtin_amdgcn_s_wait_tensorcnt)
  __builtin_amdgcn_s_wait_tensorcnt(0);
#else
  asm volatile("s_wait_tensorcnt 0x0" ::: "memory");
#endif
}

// A fragment 16x32 (MxK). Contiguous K order (not the ISA table order): lanes 0-15 hold M=lane; per-lane
// halves 0..15 = K[16g+0..15]   (g = lane>>4).
DEV v16bf load_frag_a(const u16* base, int ld, int row0, int k0, int lane) {
  const int m = lane & 15, g = lane >> 4;
  const u16* p = base + (size_t)(row0 + m) * ld + k0 + 16 * g;
  union { v16bf bf; v8u h[2]; } f;
  f.h[0] = *reinterpret_cast<const v8u*>(p);
  f.h[1] = *reinterpret_cast<const v8u*>(p + 8);
  return f.bf;
}
// B fragment 32x16 (KxN) "column-major": lane n = lane&15 holds column n,
// 16 contiguous K halves at K = 16g + 0..15 -> one 32-byte load.
DEV v16bf load_frag_b(const u16* base, int ld, int n0, int k0, int lane) {
  const int n = lane & 15, g = lane >> 4;
  const u16* p = base + (size_t)(n0 + n) * ld + k0 + 16 * g;
  union { v16bf bf; v16u u; } f;
  f.u = *reinterpret_cast<const v16u*>(p);
  return f.bf;
}

// ---------------------------------------------------------------------------
// Prep: f32 -> bf16 elementwise (vectorized 4-wide). n4 = n/4.
// ---------------------------------------------------------------------------
__global__ __launch_bounds__(256) void cvt_f32_bf16(
    const float* __restrict__ in, u16* __restrict__ out, int n4) {
  const int i = blockIdx.x * 256 + threadIdx.x;
  if (i < n4) {
    const float4 v = reinterpret_cast<const float4*>(in)[i];
    ushort4 o;
    o.x = bf16_of(v.x); o.y = bf16_of(v.y);
    o.z = bf16_of(v.z); o.w = bf16_of(v.w);
    reinterpret_cast<ushort4*>(out)[i] = o;
  }
}

// ---------------------------------------------------------------------------
// GEMM: C[M,Nout] = X[M,K]_bf16 * W[Nout,K]_bf16^T + bias_f32.
// Block tile 128x128, BK=64, double-buffered staging via TDM (wave 0 issues
// one 128x64 tile descriptor per operand; TENSORcnt fence; block barrier).
// 256 threads = 8 waves (2x4); each wave owns 64x32 (4x2 wmma accumulators).
// ---------------------------------------------------------------------------
template <typename TO>
__global__ __launch_bounds__(256) void gemm_bf16_wmma(
    const u16* __restrict__ X, const u16* __restrict__ W,
    const float* __restrict__ bias, TO* __restrict__ C,
    int M, int Nout, int K) {
  __shared__ alignas(32) u16 As[2][128 * 64];   // 2 x 16 KB
  __shared__ alignas(32) u16 Bs[2][128 * 64];   // 2 x 16 KB
  const int tid  = threadIdx.x;
  const int lane = tid & 31, wid = tid >> 5;
  const int wm = wid >> 2, wn = wid & 3;
  const int m0 = blockIdx.y * 128, n0 = blockIdx.x * 128;

  v8f acc[4][2];
#pragma unroll
  for (int i = 0; i < 4; ++i)
#pragma unroll
    for (int j = 0; j < 2; ++j) acc[i][j] = {};

  auto stage = [&](int buf, int k0) {     // one TDM descriptor per operand
    if (wid == 0) {
      tdm_load_2d_bf16(X + (size_t)m0 * K + k0,
                       (uint32_t)(uintptr_t)&As[buf][0], 64, 128, (uint32_t)K);
      tdm_load_2d_bf16(W + (size_t)n0 * K + k0,
                       (uint32_t)(uintptr_t)&Bs[buf][0], 64, 128, (uint32_t)K);
    }
  };

  stage(0, 0);
  if (wid == 0) tensor_wait0();
  __syncthreads();

  int buf = 0;
  const int KT = K >> 6;
  for (int kt = 0; kt < KT; ++kt) {
    if (kt + 1 < KT) stage(buf ^ 1, (kt + 1) * 64);   // DMA next slab
#pragma unroll
    for (int c = 0; c < 2; ++c) {                     // 16 wmma on current
      const v16bf b0 = load_frag_b(&Bs[buf][0], 64, wn * 32,      c * 32, lane);
      const v16bf b1 = load_frag_b(&Bs[buf][0], 64, wn * 32 + 16, c * 32, lane);
#pragma unroll
      for (int i = 0; i < 4; ++i) {
        const v16bf a = load_frag_a(&As[buf][0], 64, wm * 64 + i * 16, c * 32, lane);
        acc[i][0] = wmma_bf16(a, b0, acc[i][0]);
        acc[i][1] = wmma_bf16(a, b1, acc[i][1]);
      }
    }
    if (wid == 0) tensor_wait0();      // wave 0 fences its DMA
    __syncthreads();                   // then everyone may consume
    buf ^= 1;
  }

  // C/D layout: element (m,n) -> vgpr m&7, lane n + 16*(m>>3)
  const int g = lane >> 4, ln = lane & 15;
#pragma unroll
  for (int i = 0; i < 4; ++i) {
    const int mrow = m0 + wm * 64 + i * 16 + 8 * g;
#pragma unroll
    for (int j = 0; j < 2; ++j) {
      const int n = n0 + wn * 32 + j * 16 + ln;
      const float bv = bias[n];
#pragma unroll
      for (int r = 0; r < 8; ++r)
        st_out(&C[(size_t)(mrow + r) * Nout + n], acc[i][j][r] + bv);
    }
  }
}

// ---------------------------------------------------------------------------
// Flash attention. One block per (b, h, 128-query-row tile): 1024 blocks.
// 8 waves; wave w owns query rows [tile*128 + w*16, +16).  N chunk = 64.
// ---------------------------------------------------------------------------
__global__ __launch_bounds__(256) void attn_fmha(
    const u16* __restrict__ Qb, const u16* __restrict__ Kb,
    const u16* __restrict__ Vb, const unsigned char* __restrict__ mask,
    u16* __restrict__ AO) {
  constexpr int Dm = 1024, Hh = 64, Nn = 1536, Tt = 1024;
  __shared__ alignas(32) u16 Ks[64 * 64];        // [n_local][d]
  __shared__ alignas(32) u16 Vt[64 * 64];        // [d][n_local] (transposed)
  __shared__ alignas(32) u16 Ps[8][16 * 64];     // per-wave P scratch
  __shared__ float mb[64];

  const int tid = threadIdx.x, lane = tid & 31, w = tid >> 5;
  const int bx = blockIdx.x;
  const int ttile = bx & 7;
  const int h = (bx >> 3) & 15;
  const int b = bx >> 7;
  const int qbase = ttile * 128 + w * 16;
  const int g = lane >> 4, ln = lane & 15;

  // Q fragments for this wave's 16 rows, K-dim 64 = two 32-chunks (registers)
  const u16* Qh = Qb + (size_t)(b * Tt + qbase) * Dm + h * Hh;
  v16bf qa[2];
  qa[0] = load_frag_a(Qh, Dm, 0, 0,  lane);
  qa[1] = load_frag_a(Qh, Dm, 0, 32, lane);

  float rowmax[8], rowsum[8];
  v8f oacc[4];
#pragma unroll
  for (int r = 0; r < 8; ++r) { rowmax[r] = -3.0e38f; rowsum[r] = 0.0f; }
#pragma unroll
  for (int j = 0; j < 4; ++j) oacc[j] = {};

  for (int nb = 0; nb < Nn; nb += 64) {
    __syncthreads();                 // previous chunk fully consumed
    // K chunk via per-lane async DMA; V chunk transposed via VGPR path
#pragma unroll
    for (int i = 0; i < 2; ++i) {
      const int chunk = i * 256 + tid;
      const int r = chunk >> 3, c8 = (chunk & 7) * 8;
      const size_t src = (size_t)(b * Nn + nb + r) * Dm + h * Hh + c8;
      async_copy16(&Ks[r * 64 + c8], Kb + src);
      const uint4 vv = *reinterpret_cast<const uint4*>(Vb + src);
      const u16* pv = reinterpret_cast<const u16*>(&vv);
#pragma unroll
      for (int q = 0; q < 8; ++q) Vt[(c8 + q) * 64 + r] = pv[q];
    }
    if (tid < 64) mb[tid] = mask[b * Nn + nb + tid] ? -1.0e30f : 0.0f;
    async_wait0();
    __syncthreads();

    // S = (Q*K^T)*scale + maskbias, four 16x16 tiles over the 64-wide chunk
    float sv[4][8], tmax[8];
#pragma unroll
    for (int r = 0; r < 8; ++r) tmax[r] = -3.0e38f;
#pragma unroll
    for (int j = 0; j < 4; ++j) {
      v8f s = {};
#pragma unroll
      for (int c = 0; c < 2; ++c) {
        const v16bf bk = load_frag_b(Ks, 64, j * 16, c * 32, lane);
        s = wmma_bf16(qa[c], bk, s);
      }
      const float madd = mb[j * 16 + ln];
#pragma unroll
      for (int r = 0; r < 8; ++r) {
        const float v = s[r] * 0.125f + madd;   // scale = HD^-0.5 = 1/8
        sv[j][r] = v;
        tmax[r] = fmaxf(tmax[r], v);
      }
    }
    // row max across the 16 lanes of each half-group (rows r+8g)
#pragma unroll
    for (int r = 0; r < 8; ++r)
#pragma unroll
      for (int msk = 1; msk < 16; msk <<= 1)
        tmax[r] = fmaxf(tmax[r], __shfl_xor(tmax[r], msk, 32));

    float pls[8];
#pragma unroll
    for (int r = 0; r < 8; ++r) {
      const float nm = fmaxf(rowmax[r], tmax[r]);
      const float fr = __expf(rowmax[r] - nm);
      rowmax[r] = nm;
      rowsum[r] *= fr;
#pragma unroll
      for (int jj = 0; jj < 4; ++jj) oacc[jj][r] *= fr;
      pls[r] = 0.0f;
    }
    // P = exp(S - rowmax); write through per-wave LDS to re-layout C -> A
#pragma unroll
    for (int j = 0; j < 4; ++j)
#pragma unroll
      for (int r = 0; r < 8; ++r) {
        const float p = __expf(sv[j][r] - rowmax[r]);
        pls[r] += p;
        Ps[w][(r + 8 * g) * 64 + j * 16 + ln] = bf16_of(p);
      }
#pragma unroll
    for (int r = 0; r < 8; ++r) {
      float t = pls[r];
#pragma unroll
      for (int msk = 1; msk < 16; msk <<= 1) t += __shfl_xor(t, msk, 32);
      rowsum[r] += t;
    }
    // O += P * V   (K-dim = 64 local keys, two 32-chunks; four d tiles)
#pragma unroll
    for (int c = 0; c < 2; ++c) {
      const v16bf pa = load_frag_a(Ps[w], 64, 0, c * 32, lane);
#pragma unroll
      for (int jj = 0; jj < 4; ++jj) {
        const v16bf bvf = load_frag_b(Vt, 64, jj * 16, c * 32, lane);
        oacc[jj] = wmma_bf16(pa, bvf, oacc[jj]);
      }
    }
  }

  // normalize and store bf16 attention output [B,T,D] at column h*64
#pragma unroll
  for (int r = 0; r < 8; ++r) rowsum[r] = 1.0f / rowsum[r];
  const size_t obase = (size_t)(b * Tt + qbase) * Dm + h * Hh;
#pragma unroll
  for (int jj = 0; jj < 4; ++jj)
#pragma unroll
    for (int r = 0; r < 8; ++r) {
      const int row = r + 8 * g;
      AO[obase + (size_t)row * Dm + jj * 16 + ln] = bf16_of(oacc[jj][r] * rowsum[r]);
    }
}

// ---------------------------------------------------------------------------
extern "C" void kernel_launch(void* const* d_in, const int* in_sizes, int n_in,
                              void* d_out, int out_size, void* d_ws, size_t ws_size,
                              hipStream_t stream) {
  (void)in_sizes; (void)n_in; (void)out_size; (void)ws_size;
  const float* query     = (const float*)d_in[0];
  const float* key_value = (const float*)d_in[1];
  const unsigned char* mask = (const unsigned char*)d_in[2];
  const float* Wq = (const float*)d_in[3];
  const float* bq = (const float*)d_in[4];
  const float* Wk = (const float*)d_in[5];
  const float* bk = (const float*)d_in[6];
  const float* Wv = (const float*)d_in[7];
  const float* bv = (const float*)d_in[8];
  const float* Wo = (const float*)d_in[9];
  const float* bo = (const float*)d_in[10];
  float* out = (float*)d_out;

  const size_t nQ = (size_t)8 * 1024 * 1024;   // 8.4M
  const size_t nKV = (size_t)8 * 1536 * 1024;  // 12.6M
  const size_t nW = (size_t)1024 * 1024;

  u16* Xq  = (u16*)d_ws;
  u16* Xkv = Xq  + nQ;
  u16* Wqb = Xkv + nKV;
  u16* Wkb = Wqb + nW;
  u16* Wvb = Wkb + nW;
  u16* Wob = Wvb + nW;
  u16* Qb  = Wob + nW;
  u16* Kb  = Qb  + nQ;
  u16* Vb  = Kb  + nKV;
  u16* AO  = Vb  + nKV;

  dim3 blk(256);
  // prep: f32 -> bf16
  cvt_f32_bf16<<<dim3((unsigned)(nQ  / 4 / 256)), blk, 0, stream>>>(query,     Xq,  (int)(nQ  / 4));
  cvt_f32_bf16<<<dim3((unsigned)(nKV / 4 / 256)), blk, 0, stream>>>(key_value, Xkv, (int)(nKV / 4));
  cvt_f32_bf16<<<dim3((unsigned)(nW  / 4 / 256)), blk, 0, stream>>>(Wq, Wqb, (int)(nW / 4));
  cvt_f32_bf16<<<dim3((unsigned)(nW  / 4 / 256)), blk, 0, stream>>>(Wk, Wkb, (int)(nW / 4));
  cvt_f32_bf16<<<dim3((unsigned)(nW  / 4 / 256)), blk, 0, stream>>>(Wv, Wvb, (int)(nW / 4));
  cvt_f32_bf16<<<dim3((unsigned)(nW  / 4 / 256)), blk, 0, stream>>>(Wo, Wob, (int)(nW / 4));
  // projections
  gemm_bf16_wmma<u16><<<dim3(8, 64), blk, 0, stream>>>(Xq,  Wqb, bq, Qb, 8192,  1024, 1024);
  gemm_bf16_wmma<u16><<<dim3(8, 96), blk, 0, stream>>>(Xkv, Wkb, bk, Kb, 12288, 1024, 1024);
  gemm_bf16_wmma<u16><<<dim3(8, 96), blk, 0, stream>>>(Xkv, Wvb, bv, Vb, 12288, 1024, 1024);
  // attention
  attn_fmha<<<dim3(1024), blk, 0, stream>>>(Qb, Kb, Vb, mask, AO);
  // output projection
  gemm_bf16_wmma<float><<<dim3(8, 64), blk, 0, stream>>>(AO, Wob, bo, out, 8192, 1024, 1024);
}
